// PerformerAttention_37778532335874
// MI455X (gfx1250) — compile-verified, matrix operand order checked by reading, not run on hardware
//
#include <hip/hip_runtime.h>
#include <hip/hip_bf16.h>
#include <cstdint>

typedef __bf16 bf16;
typedef __attribute__((ext_vector_type(16))) __bf16 v16bf;
typedef __attribute__((ext_vector_type(8)))  float  v8f;
typedef __attribute__((ext_vector_type(16))) float  v16f;

// Problem constants
constexpr int Bc = 2, Lc = 8192, Dc = 1024, Hc = 16, DHc = 64;
constexpr int Mfeat = 266;          // real feature count
constexpr int Mp    = 288;          // padded to multiple of 32 (K-loop friendly)
constexpr int BL    = Bc * Lc;      // 16384 rows
constexpr int NV    = 80;           // DH + 16 (ones column at 64, zeros after)
constexpr float QK_SCALE = 0.17677669529663687f;   // D^-0.25
constexpr float NC       = 0.06131393394849658f;   // 266^-0.5
constexpr float KEPS     = 1e-4f;
constexpr float NSTAB    = 1e-6f;

__device__ __forceinline__ v8f wmma_bf16(v16bf a, v16bf b, v8f c) {
  return __builtin_amdgcn_wmma_f32_16x16x32_bf16(false, a, false, b, (short)0, c, false, false);
}

// ---------------- elementwise converters ----------------
__global__ void cvt_f32_bf16(const float* __restrict__ in, bf16* __restrict__ out, int n) {
  int i = blockIdx.x * blockDim.x + threadIdx.x;
  if (i < n) out[i] = (bf16)in[i];
}

__global__ void build_wf(const float* __restrict__ wf, bf16* __restrict__ out) {
  int i = blockIdx.x * blockDim.x + threadIdx.x;
  if (i >= Mp * DHc) return;
  int m = i / DHc, d = i % DHc;
  out[i] = (m < Mfeat) ? (bf16)wf[m * DHc + d] : (bf16)0.0f;
}

__global__ void init_stab(unsigned* p) { *p = 0xFF800000u; }  // bits(-inf)

// ---------------- C = X @ W^T + bias   (X:[rows,K] bf16, W:[N,K] bf16) ----------------
// Block = 128 threads (4 waves, 2x2), each wave owns a 32x32 tile -> 64x64 per WG.
template <bool OUT_BF16>
__global__ __launch_bounds__(128) void gemm_xwt(const bf16* __restrict__ X,
                                                const bf16* __restrict__ W,
                                                const float* __restrict__ bias,
                                                void* __restrict__ out,
                                                int K, int N, float scale) {
  const int lane = threadIdx.x & 31;
  const int wave = threadIdx.x >> 5;
  const int lr = lane & 15, lh = lane >> 4;
  const int m0 = blockIdx.y * 64 + (wave >> 1) * 32;
  const int n0 = blockIdx.x * 64 + (wave & 1) * 32;

  v8f c00 = {}, c01 = {}, c10 = {}, c11 = {};
  const bf16* ap0 = X + (size_t)(m0 + lr) * K + lh * 16;
  const bf16* ap1 = ap0 + (size_t)16 * K;
  const bf16* bp0 = W + (size_t)(n0 + lr) * K + lh * 16;
  const bf16* bp1 = bp0 + (size_t)16 * K;

  for (int kk = 0; kk < K; kk += 32) {
    __builtin_prefetch(ap0 + kk + 128, 0, 1);
    __builtin_prefetch(bp0 + kk + 128, 0, 1);
    v16bf a0 = *(const v16bf*)(ap0 + kk);
    v16bf a1 = *(const v16bf*)(ap1 + kk);
    v16bf b0 = *(const v16bf*)(bp0 + kk);
    v16bf b1 = *(const v16bf*)(bp1 + kk);
    c00 = wmma_bf16(a0, b0, c00);
    c01 = wmma_bf16(a0, b1, c01);
    c10 = wmma_bf16(a1, b0, c10);
    c11 = wmma_bf16(a1, b1, c11);
  }

  const float bia = bias[n0 + lr];
  const float bib = bias[n0 + 16 + lr];
#pragma unroll
  for (int v = 0; v < 8; v++) {
    int r0 = m0 + v + lh * 8;
    int r1 = r0 + 16;
    int ca = n0 + lr, cb = n0 + 16 + lr;
    float e00 = (c00[v] + bia) * scale;
    float e01 = (c01[v] + bib) * scale;
    float e10 = (c10[v] + bia) * scale;
    float e11 = (c11[v] + bib) * scale;
    if (OUT_BF16) {
      bf16* o = (bf16*)out;
      o[(size_t)r0 * N + ca] = (bf16)e00;  o[(size_t)r0 * N + cb] = (bf16)e01;
      o[(size_t)r1 * N + ca] = (bf16)e10;  o[(size_t)r1 * N + cb] = (bf16)e11;
    } else {
      float* o = (float*)out;
      o[(size_t)r0 * N + ca] = e00;  o[(size_t)r0 * N + cb] = e01;
      o[(size_t)r1 * N + ca] = e10;  o[(size_t)r1 * N + cb] = e11;
    }
  }
}

// ---------------- h_k = -0.5*||k_row_head||^2 ; global stab = max(h_k) ----------------
__global__ __launch_bounds__(256) void rowstats(const bf16* __restrict__ Kp,
                                                float* __restrict__ hk,
                                                unsigned* __restrict__ stab_bits) {
  __shared__ float sm[256];
  int i = blockIdx.x * 256 + threadIdx.x;        // (b*L+l)*H + h, exact grid
  int row = i / Hc, h = i % Hc;
  const bf16* p = Kp + (size_t)row * Dc + h * DHc;
  float s = 0.f;
#pragma unroll
  for (int d = 0; d < DHc; d++) { float x = (float)p[d]; s += x * x; }
  float hv = -0.5f * s;
  hk[i] = hv;
  sm[threadIdx.x] = hv;
  __syncthreads();
  for (int off = 128; off > 0; off >>= 1) {
    if (threadIdx.x < off) sm[threadIdx.x] = fmaxf(sm[threadIdx.x], sm[threadIdx.x + off]);
    __syncthreads();
  }
  // h_k <= 0 always: max(float) == min(uint bit pattern) over negatives
  if (threadIdx.x == 0) atomicMin(stab_bits, __float_as_uint(sm[0]));
}

// ---------------- feature projection + exp kernelization ----------------
// q-side (KSIDE=0): qp[b,h,l,m] = nc*(exp(proj)+eps)               -> layout [BH, L, Mp]
// k-side (KSIDE=1): kT[b,h,m,l] = nc*(exp(h_k+proj-stab)+eps)*mask -> layout [BH, Mp, L]
template <bool KSIDE>
__global__ __launch_bounds__(32) void feat_exp(const bf16* __restrict__ QK,
                                               const bf16* __restrict__ Wf,
                                               const float* __restrict__ hk,
                                               const unsigned* __restrict__ stab_bits,
                                               const float* __restrict__ mask,
                                               bf16* __restrict__ outP) {
  const int lane = threadIdx.x & 31, lr = lane & 15, lh = lane >> 4;
  const int m0 = blockIdx.x * 16;
  const int l0 = blockIdx.y * 16;
  const int bh = blockIdx.z;
  const int b = bh >> 4, h = bh & 15;

  v8f c = {};
  const bf16* ap = QK + (size_t)(b * Lc + l0 + lr) * Dc + h * DHc + lh * 16;
  const bf16* bp = Wf + (size_t)(m0 + lr) * DHc + lh * 16;
#pragma unroll
  for (int d = 0; d < DHc; d += 32) {
    v16bf a = *(const v16bf*)(ap + d);
    v16bf bb = *(const v16bf*)(bp + d);
    c = wmma_bf16(a, bb, c);
  }

  float stab = 0.0f;
  if (KSIDE) stab = __uint_as_float(*stab_bits);
  const int m = m0 + lr;

#pragma unroll
  for (int v = 0; v < 8; v++) {
    int l = l0 + v + lh * 8;
    float val = 0.0f;
    if (m < Mfeat) {
      if (KSIDE) {
        float hv = hk[(b * Lc + l) * Hc + h];
        val = NC * (expf(hv + c[v] - stab) + KEPS);
        val *= mask[b * Lc + l];
      } else {
        val = NC * (expf(c[v]) + KEPS);
      }
    }
    if (KSIDE) outP[((size_t)bh * Mp + m) * Lc + l] = (bf16)val;   // transposed store
    else       outP[((size_t)bh * Lc + l) * Mp + m] = (bf16)val;
  }
}

// ---------------- v^T with ones column:  vT[bh, n, l] ----------------
__global__ void build_vt(const bf16* __restrict__ V, bf16* __restrict__ vT) {
  int i = blockIdx.x * blockDim.x + threadIdx.x;
  if (i >= Bc * Hc * NV * Lc) return;
  int l = i % Lc; int t = i / Lc; int n = t % NV; int bh = t / NV;
  int b = bh >> 4, h = bh & 15;
  bf16 val;
  if (n < DHc)       val = V[(size_t)(b * Lc + l) * Dc + h * DHc + n];
  else if (n == DHc) val = (bf16)1.0f;
  else               val = (bf16)0.0f;
  vT[i] = val;
}

// ---------------- kv^T[bh, n, m] = sum_l vT[n,l] * k'[l,m]  (K = L = 8192) ----------------
// Double-buffered global->LDS async copy (ASYNCcnt) feeding the WMMA stream.
// Each lane stages its own 32B operand slice: two async b128s per operand; the
// single `offset:16` advances both the LDS and global addresses (ISA 8.4).
__global__ __launch_bounds__(32) void kv_gemm(const bf16* __restrict__ vT,
                                              const bf16* __restrict__ kT,
                                              float* __restrict__ kvt) {
  const int lane = threadIdx.x & 31, lr = lane & 15, lh = lane >> 4;
  const int m0 = blockIdx.x * 16;   // feature dim
  const int n0 = blockIdx.y * 16;   // value dim (NV)
  const int bh = blockIdx.z;

  __shared__ bf16 sa[2][512];       // 2 stages x (32 lanes x 16 bf16)
  __shared__ bf16 sb[2][512];

  const bf16* ap = vT + (size_t)(bh * NV + n0 + lr) * Lc + lh * 16;
  const bf16* bp = kT + (size_t)(bh * Mp + m0 + lr) * Lc + lh * 16;

  // LDS byte offsets (low 32 bits of the flat shared address = LDS offset)
  uint32_t la[2], lb[2];
  la[0] = (uint32_t)(uintptr_t)(&sa[0][lane * 16]);
  la[1] = (uint32_t)(uintptr_t)(&sa[1][lane * 16]);
  lb[0] = (uint32_t)(uintptr_t)(&sb[0][lane * 16]);
  lb[1] = (uint32_t)(uintptr_t)(&sb[1][lane * 16]);

  auto issue = [&](int stage, int l) {
    uint64_t ga = (uint64_t)(uintptr_t)(ap + l);
    uint64_t gb = (uint64_t)(uintptr_t)(bp + l);
    asm volatile(
        "global_load_async_to_lds_b128 %0, %2, off\n\t"
        "global_load_async_to_lds_b128 %0, %2, off offset:16\n\t"
        "global_load_async_to_lds_b128 %1, %3, off\n\t"
        "global_load_async_to_lds_b128 %1, %3, off offset:16"
        :: "v"(la[stage]), "v"(lb[stage]), "v"(ga), "v"(gb)
        : "memory");
  };

  v8f c = {};
  issue(0, 0);
  for (int l = 0; l < Lc; l += 32) {
    const int st = (l >> 5) & 1;
    if (l + 32 < Lc) {
      issue(st ^ 1, l + 32);
      // async loads complete in order: <=4 outstanding => current stage landed
      asm volatile("s_wait_asynccnt 0x4" ::: "memory");
    } else {
      asm volatile("s_wait_asynccnt 0x0" ::: "memory");
    }
    v16bf a = *(const v16bf*)(&sa[st][lane * 16]);
    v16bf b = *(const v16bf*)(&sb[st][lane * 16]);
    c = wmma_bf16(a, b, c);
  }

#pragma unroll
  for (int v = 0; v < 8; v++) {
    int n = n0 + v + lh * 8;
    kvt[((size_t)bh * NV + n) * Mp + m0 + lr] = c[v];
  }
}

// ---------------- E[bh, l, n] = sum_m q'[l,m] * kv[m,n]  (K = Mp = 288) ----------------
__global__ __launch_bounds__(32) void out_gemm(const bf16* __restrict__ qp,
                                               const float* __restrict__ kvt,
                                               float* __restrict__ E) {
  const int lane = threadIdx.x & 31, lr = lane & 15, lh = lane >> 4;
  const int n0 = blockIdx.x * 16;   // NV
  const int l0 = blockIdx.y * 16;
  const int bh = blockIdx.z;
  v8f c = {};
  const bf16*  ap = qp  + (size_t)(bh * Lc + l0 + lr) * Mp + lh * 16;
  const float* bp = kvt + (size_t)(bh * NV + n0 + lr) * Mp + lh * 16;
#pragma unroll
  for (int m = 0; m < Mp; m += 32) {
    v16bf a = *(const v16bf*)(ap + m);
    v16f  bf_ = *(const v16f*)(bp + m);
    v16bf bb;
#pragma unroll
    for (int t = 0; t < 16; t++) bb[t] = (bf16)bf_[t];
    c = wmma_bf16(a, bb, c);
  }
#pragma unroll
  for (int v = 0; v < 8; v++) {
    int l = l0 + v + lh * 8;
    E[((size_t)bh * Lc + l) * NV + n0 + lr] = c[v];
  }
}

// ---------------- ctx = E[:, :64] / denom(E[:,64]) ----------------
__global__ void div_ctx(const float* __restrict__ E, bf16* __restrict__ ctx) {
  int i = blockIdx.x * blockDim.x + threadIdx.x;
  if (i >= Bc * Hc * Lc * DHc) return;
  int n = i & 63; int t = i >> 6; int l = t % Lc; int bh = t / Lc;
  int b = bh >> 4, h = bh & 15;
  const float* e = E + ((size_t)bh * Lc + l) * NV;
  float d = e[DHc];
  d = d + 2.0f * NSTAB * ((fabsf(d) <= NSTAB) ? 1.0f : 0.0f);
  ctx[(size_t)(b * Lc + l) * Dc + h * DHc + n] = (bf16)(e[n] / d);
}

// ======================================================================
static inline char* align_up(char* p, size_t a) {
  return (char*)(((uintptr_t)p + a - 1) & ~(uintptr_t)(a - 1));
}

extern "C" void kernel_launch(void* const* d_in, const int* in_sizes, int n_in,
                              void* d_out, int out_size, void* d_ws, size_t ws_size,
                              hipStream_t stream) {
  (void)in_sizes; (void)n_in; (void)out_size; (void)ws_size;
  const float* q_in = (const float*)d_in[0];
  const float* k_in = (const float*)d_in[1];
  const float* v_in = (const float*)d_in[2];
  const float* mask = (const float*)d_in[3];
  const float* Wq = (const float*)d_in[4];  const float* bq = (const float*)d_in[5];
  const float* Wk = (const float*)d_in[6];  const float* bk = (const float*)d_in[7];
  const float* Wv = (const float*)d_in[8];  const float* bv = (const float*)d_in[9];
  const float* Wo = (const float*)d_in[10]; const float* bo = (const float*)d_in[11];
  const float* Wf = (const float*)d_in[12];
  float* out = (float*)d_out;

  char* p = (char*)d_ws;
  auto alloc = [&](size_t bytes) -> char* { p = align_up(p, 256); char* r = p; p += bytes; return r; };

  bf16* xq_b = (bf16*)alloc((size_t)BL * Dc * 2);
  bf16* xk_b = (bf16*)alloc((size_t)BL * Dc * 2);
  bf16* xv_b = (bf16*)alloc((size_t)BL * Dc * 2);
  bf16* wq_b = (bf16*)alloc((size_t)Dc * Dc * 2);
  bf16* wk_b = (bf16*)alloc((size_t)Dc * Dc * 2);
  bf16* wv_b = (bf16*)alloc((size_t)Dc * Dc * 2);
  bf16* wo_b = (bf16*)alloc((size_t)Dc * Dc * 2);
  bf16* wf_b = (bf16*)alloc((size_t)Mp * DHc * 2);
  bf16* Qb   = (bf16*)alloc((size_t)BL * Dc * 2);
  bf16* Kb   = (bf16*)alloc((size_t)BL * Dc * 2);
  bf16* Vb   = (bf16*)alloc((size_t)BL * Dc * 2);
  float* hk  = (float*)alloc((size_t)BL * Hc * 4);
  unsigned* stab = (unsigned*)alloc(256);
  bf16* qp   = (bf16*)alloc((size_t)Bc * Hc * Lc * Mp * 2);
  bf16* kT   = (bf16*)alloc((size_t)Bc * Hc * Mp * Lc * 2);
  bf16* vT   = (bf16*)alloc((size_t)Bc * Hc * NV * Lc * 2);
  float* kvt = (float*)alloc((size_t)Bc * Hc * NV * Mp * 4);
  float* E   = (float*)alloc((size_t)Bc * Hc * Lc * NV * 4);
  bf16* ctx  = (bf16*)alloc((size_t)BL * Dc * 2);

  const int nBD = BL * Dc, nDD = Dc * Dc;
  cvt_f32_bf16<<<(nBD + 255) / 256, 256, 0, stream>>>(q_in, xq_b, nBD);
  cvt_f32_bf16<<<(nBD + 255) / 256, 256, 0, stream>>>(k_in, xk_b, nBD);
  cvt_f32_bf16<<<(nBD + 255) / 256, 256, 0, stream>>>(v_in, xv_b, nBD);
  cvt_f32_bf16<<<(nDD + 255) / 256, 256, 0, stream>>>(Wq, wq_b, nDD);
  cvt_f32_bf16<<<(nDD + 255) / 256, 256, 0, stream>>>(Wk, wk_b, nDD);
  cvt_f32_bf16<<<(nDD + 255) / 256, 256, 0, stream>>>(Wv, wv_b, nDD);
  cvt_f32_bf16<<<(nDD + 255) / 256, 256, 0, stream>>>(Wo, wo_b, nDD);
  build_wf<<<(Mp * DHc + 255) / 256, 256, 0, stream>>>(Wf, wf_b);

  dim3 ggrid(Dc / 64, BL / 64);
  gemm_xwt<true><<<ggrid, 128, 0, stream>>>(xq_b, wq_b, bq, Qb, Dc, Dc, QK_SCALE);
  gemm_xwt<true><<<ggrid, 128, 0, stream>>>(xk_b, wk_b, bk, Kb, Dc, Dc, QK_SCALE);
  gemm_xwt<true><<<ggrid, 128, 0, stream>>>(xv_b, wv_b, bv, Vb, Dc, Dc, 1.0f);

  init_stab<<<1, 1, 0, stream>>>(stab);
  rowstats<<<(BL * Hc) / 256, 256, 0, stream>>>(Kb, hk, stab);

  dim3 fgrid(Mp / 16, Lc / 16, Bc * Hc);
  feat_exp<false><<<fgrid, 32, 0, stream>>>(Qb, wf_b, nullptr, nullptr, nullptr, qp);
  feat_exp<true><<<fgrid, 32, 0, stream>>>(Kb, wf_b, hk, stab, mask, kT);

  int nvt = Bc * Hc * NV * Lc;
  build_vt<<<(nvt + 255) / 256, 256, 0, stream>>>(Vb, vT);

  kv_gemm<<<dim3(Mp / 16, NV / 16, Bc * Hc), 32, 0, stream>>>(vT, kT, kvt);
  out_gemm<<<dim3(NV / 16, Lc / 16, Bc * Hc), 32, 0, stream>>>(qp, kvt, E);

  div_ctx<<<(BL * Dc + 255) / 256, 256, 0, stream>>>(E, ctx);

  gemm_xwt<false><<<ggrid, 128, 0, stream>>>(ctx, wo_b, bo, out, Dc, Dc, 1.0f);
}
